// ContrastiveLoss_13761075216965
// MI455X (gfx1250) — matrix operand order repaired and checked by reading, not compiled, not run on hardware
//
#include <hip/hip_runtime.h>

typedef __attribute__((ext_vector_type(16))) _Float16 v16h;
typedef __attribute__((ext_vector_type(8)))  _Float16 v8h;
typedef __attribute__((ext_vector_type(8)))  float    v8f;

#define TEMP_INV 14.285714285714286f   /* 1 / 0.07 */
#define EPS_F    1e-8f

constexpr int NB = 8192;          // batch
constexpr int ND = 512;           // feature dim
constexpr int NTILES = NB / 16;   // 512 column tiles
constexpr int MROWS = 32;         // rows per workgroup (2 x 16-row WMMA tiles)
constexpr int NMBLK = NB / MROWS; // 256 workgroups

// ---------------------------------------------------------------------------
// Kernel 1: L2-normalize rows, fp32 -> f16 (matches ref: norm clamped at 1e-12)
// ---------------------------------------------------------------------------
__global__ __launch_bounds__(256) void norm_rows(const float* __restrict__ feat,
                                                 _Float16* __restrict__ out) {
    __shared__ float red[8];
    __shared__ float s_inv;
    const int row = blockIdx.x;
    const int t = threadIdx.x;
    const float* rp = feat + (size_t)row * ND;
    float v0 = rp[t];
    float v1 = rp[t + 256];
    float ss = v0 * v0 + v1 * v1;
#pragma unroll
    for (int m = 16; m >= 1; m >>= 1) ss += __shfl_xor(ss, m, 32);
    if ((t & 31) == 0) red[t >> 5] = ss;
    __syncthreads();
    if (t == 0) {
        float s = 0.f;
#pragma unroll
        for (int i = 0; i < 8; ++i) s += red[i];
        float n = sqrtf(s);
        n = fmaxf(n, 1e-12f);
        s_inv = 1.0f / n;
    }
    __syncthreads();
    const float inv = s_inv;
    _Float16* op = out + (size_t)row * ND;
    op[t]       = (_Float16)(v0 * inv);
    op[t + 256] = (_Float16)(v1 * inv);
}

// ---------------------------------------------------------------------------
// Kernel 2: fused sim = f f^T / T, exp, mask, row sums, per-block loss partials
// One workgroup (8 waves) per 32-row M-block. Every wave holds BOTH 16-row
// A fragments for the full K=512 in registers (2 x 128 VGPRs); each B fragment
// load feeds two WMMAs (two independent accumulator chains).
// Wave w covers column tiles jt = w, w+8, ...
// ---------------------------------------------------------------------------
__global__ __launch_bounds__(256, 1) void supcon_main(const _Float16* __restrict__ F,
                                                      const int* __restrict__ labels,
                                                      float* __restrict__ partial) {
    __shared__ float s_pos[8][MROWS];
    __shared__ float s_neg[8][MROWS];

    const int r0   = blockIdx.x * MROWS;
    const int lane = threadIdx.x & 31;
    const int wid  = threadIdx.x >> 5;
    const int lo   = lane & 15;
    const int hi   = lane >> 4;

    // ---- Load A fragments: 2 row-tiles x 16 K-steps (K=512, 32 per WMMA) ----
    // A 16-bit 16x32 layout: lane (l<16) = row M=l; elems 0..7 -> K=8*hi+0..7,
    // elems 8..15 -> K=16+8*hi+0..7  (ISA 7.12.2)
    v16h a[2][16];
#pragma unroll
    for (int tt = 0; tt < 2; ++tt) {
        const _Float16* arow = F + (size_t)(r0 + tt * 16 + lo) * ND;
#pragma unroll
        for (int kk = 0; kk < 16; ++kk) {
            const int k0 = kk * 32;
            v8h a0 = *(const v8h*)(arow + k0 + 8 * hi);
            v8h a1 = *(const v8h*)(arow + k0 + 16 + 8 * hi);
#pragma unroll
            for (int i = 0; i < 8; ++i) { a[tt][kk][i] = a0[i]; a[tt][kk][8 + i] = a1[i]; }
        }
    }

    // Row labels for the 16 C-rows this lane owns (rows tt*16 + i + 8*hi)
    int labrow[2][8];
#pragma unroll
    for (int tt = 0; tt < 2; ++tt)
#pragma unroll
        for (int i = 0; i < 8; ++i) labrow[tt][i] = labels[r0 + tt * 16 + i + 8 * hi];

    float pos[2][8], neg[2][8];
#pragma unroll
    for (int tt = 0; tt < 2; ++tt)
#pragma unroll
        for (int i = 0; i < 8; ++i) { pos[tt][i] = 0.f; neg[tt][i] = 0.f; }

    for (int jt = wid; jt < NTILES; jt += 8) {
        const int col  = jt * 16 + lo;
        const int labc = labels[col];
        // B 16-bit 32x16 layout: lane (l<16) = col N=l; elems 0..7 -> K=8*hi+0..7, elems 8..15 -> K=16+8*hi+0..7
        const _Float16* brow = F + (size_t)col * ND;
        v8f c0 = {}, c1 = {};
#pragma unroll
        for (int kk = 0; kk < 16; ++kk) {
            const int k0 = kk * 32 + 8 * hi;
            v8h b0 = *(const v8h*)(brow + k0);
            v8h b1 = *(const v8h*)(brow + k0 + 16);
            v16h bb;
#pragma unroll
            for (int i = 0; i < 8; ++i) { bb[i] = b0[i]; bb[8 + i] = b1[i]; }
            c0 = __builtin_amdgcn_wmma_f32_16x16x32_f16(false, a[0][kk], false, bb,
                                                        (short)0, c0, false, false);
            c1 = __builtin_amdgcn_wmma_f32_16x16x32_f16(false, a[1][kk], false, bb,
                                                        (short)0, c1, false, false);
        }
        // Epilogue: C layout (f32 16x16): VGPR i, lane l -> row i+8*hi, col lo
#pragma unroll
        for (int tt = 0; tt < 2; ++tt) {
#pragma unroll
            for (int i = 0; i < 8; ++i) {
                const float sim = (tt == 0 ? c0[i] : c1[i]) * TEMP_INV;
                const int row = r0 + tt * 16 + i + 8 * hi;
                float e = __expf(sim);
                e = (col == row) ? 0.f : e;              // zero the diagonal
                if (labc == labrow[tt][i]) pos[tt][i] += e;
                else                       neg[tt][i] += e;
            }
        }
    }

    // Reduce over the 16 columns within each half-wave (xor 1,2,4,8 stays in half)
#pragma unroll
    for (int tt = 0; tt < 2; ++tt) {
#pragma unroll
        for (int i = 0; i < 8; ++i) {
#pragma unroll
            for (int m = 8; m >= 1; m >>= 1) {
                pos[tt][i] += __shfl_xor(pos[tt][i], m, 32);
                neg[tt][i] += __shfl_xor(neg[tt][i], m, 32);
            }
        }
    }
    if (lo == 0) {
#pragma unroll
        for (int tt = 0; tt < 2; ++tt) {
#pragma unroll
            for (int i = 0; i < 8; ++i) {
                s_pos[wid][tt * 16 + i + 8 * hi] = pos[tt][i];
                s_neg[wid][tt * 16 + i + 8 * hi] = neg[tt][i];
            }
        }
    }
    __syncthreads();

    // Cross-wave reduce + per-row loss (threads 0..31 each own one row)
    if (threadIdx.x < MROWS) {
        float p = 0.f, n = 0.f;
#pragma unroll
        for (int w = 0; w < 8; ++w) { p += s_pos[w][threadIdx.x]; n += s_neg[w][threadIdx.x]; }
        s_pos[0][threadIdx.x] = -__logf(p / (p + n + EPS_F));
    }
    __syncthreads();
    if (threadIdx.x == 0) {
        float s = 0.f;
#pragma unroll
        for (int i = 0; i < MROWS; ++i) s += s_pos[0][i];
        partial[blockIdx.x] = s;
    }
}

// ---------------------------------------------------------------------------
// Kernel 3: deterministic reduction of 256 partials -> mean loss
// ---------------------------------------------------------------------------
__global__ __launch_bounds__(256) void final_reduce(const float* __restrict__ partial,
                                                    float* __restrict__ out) {
    __shared__ float red[8];
    const int t = threadIdx.x;
    float v = partial[t];
#pragma unroll
    for (int m = 16; m >= 1; m >>= 1) v += __shfl_xor(v, m, 32);
    if ((t & 31) == 0) red[t >> 5] = v;
    __syncthreads();
    if (t == 0) {
        float s = 0.f;
#pragma unroll
        for (int i = 0; i < 8; ++i) s += red[i];
        out[0] = s * (1.0f / (float)NB);
    }
}

extern "C" void kernel_launch(void* const* d_in, const int* in_sizes, int n_in,
                              void* d_out, int out_size, void* d_ws, size_t ws_size,
                              hipStream_t stream) {
    (void)in_sizes; (void)n_in; (void)out_size; (void)ws_size;
    const float* feat   = (const float*)d_in[0];
    const int*   labels = (const int*)d_in[1];
    float*       out    = (float*)d_out;

    _Float16* fnorm   = (_Float16*)d_ws;                                   // 8 MB
    float*    partial = (float*)((char*)d_ws + (size_t)NB * ND * sizeof(_Float16));

    norm_rows   <<<NB,    256, 0, stream>>>(feat, fnorm);
    supcon_main <<<NMBLK, 256, 0, stream>>>(fnorm, labels, partial);
    final_reduce<<<1,     256, 0, stream>>>(partial, out);
}
